// CausalSelfAttention_31490700214521
// MI455X (gfx1250) — compile-verified, matrix operand order checked by reading, not run on hardware
//
#include <hip/hip_runtime.h>
#include <hip/hip_bf16.h>

// ---------------------------------------------------------------------------
// Causal self-attention, B=4 S=2048 D=1024, fp32 in/out.
// bf16 WMMA (v_wmma_f32_16x16x32_bf16) for all GEMMs, fp32 accumulate.
// Score matrix materialized in bf16 (32MB, L2-resident on MI455X 192MB L2).
// Pipeline: cvt -> fused QKV NT-GEMM (V stored transposed) -> score NT-GEMM
//           -> per-row max/sumexp stats -> in-place normalize -> PV NT-GEMM.
// All GEMM wave tiles are 32x64 (8 WMMA per k-step, 1.5 b128 loads / WMMA).
// ---------------------------------------------------------------------------

typedef __bf16 bf16_t;
typedef bf16_t v16bf __attribute__((ext_vector_type(16)));
typedef bf16_t v8bf  __attribute__((ext_vector_type(8)));
typedef bf16_t v2bf  __attribute__((ext_vector_type(2)));
typedef float  v8f   __attribute__((ext_vector_type(8)));

#define BB 4
#define SS 2048
#define DD 1024

// --------------------------- 1) fp32 -> bf16 -------------------------------
__global__ void cvt_f32_bf16(const float* __restrict__ src,
                             bf16_t* __restrict__ dst, int n) {
  int i = blockIdx.x * blockDim.x + threadIdx.x;
  if (i < n) dst[i] = (bf16_t)src[i];
}

// --------------------------- 2) fused QKV GEMM -----------------------------
// C[m,n] = sum_k xb[m,k] * W[n,k] + bias[n].  Wave -> 32x64 tile.
// Q,K row-major bf16 [B*S, D]; V stored transposed Vt[B, D, S].
__global__ void __launch_bounds__(256) qkv_gemm(
    const bf16_t* __restrict__ xb,
    const bf16_t* __restrict__ Wqb, const bf16_t* __restrict__ Wkb,
    const bf16_t* __restrict__ Wvb,
    const float* __restrict__ bq, const float* __restrict__ bk,
    const float* __restrict__ bv,
    bf16_t* __restrict__ Qb, bf16_t* __restrict__ Kb,
    bf16_t* __restrict__ Vt) {
  const int lane = threadIdx.x & 31;
  const int wid  = blockIdx.x * 8 + (threadIdx.x >> 5);
  // 3 proj * 256 row-blocks(32) * 16 col-groups(64) = 12288 wave tasks
  const int ng   = wid & 15;
  const int mt   = (wid >> 4) & 255;
  const int proj = wid >> 12;
  if (proj >= 3) return;

  const bf16_t* W    = (proj == 0) ? Wqb : (proj == 1) ? Wkb : Wvb;
  const float*  bias = (proj == 0) ? bq  : (proj == 1) ? bk  : bv;

  const int m0 = mt * 32, n0 = ng * 64;
  const int lr = lane & 15, lh = lane >> 4;

  const bf16_t* aptr = xb + (size_t)(m0 + lr) * DD + lh * 16;
  const bf16_t* bptr = W  + (size_t)(n0 + lr) * DD + lh * 16;

  v8f acc[2][4] = {};
  #pragma unroll 4
  for (int kk = 0; kk < DD; kk += 32) {
    v16bf a0 = *(const v16bf*)(aptr + kk);
    v16bf a1 = *(const v16bf*)(aptr + (size_t)16 * DD + kk);
    __builtin_prefetch(aptr + kk + 128, 0, 3);   // WGP-scope prefetch
    #pragma unroll
    for (int j = 0; j < 4; ++j) {
      v16bf bf = *(const v16bf*)(bptr + (size_t)j * 16 * DD + kk);
      acc[0][j] = __builtin_amdgcn_wmma_f32_16x16x32_bf16(
          false, a0, false, bf, (short)0, acc[0][j], false, false);
      acc[1][j] = __builtin_amdgcn_wmma_f32_16x16x32_bf16(
          false, a1, false, bf, (short)0, acc[1][j], false, false);
    }
  }

  bf16_t* Out = (proj == 0) ? Qb : Kb;
  #pragma unroll
  for (int mi = 0; mi < 2; ++mi)
    #pragma unroll
    for (int j = 0; j < 4; ++j) {
      const int col = n0 + j * 16 + lr;
      const float bc = bias[col];
      #pragma unroll
      for (int i = 0; i < 8; ++i) {
        const int row = m0 + mi * 16 + i + 8 * lh;  // C layout: VGPR i + half
        const float v = acc[mi][j][i] + bc;
        if (proj < 2) {
          Out[(size_t)row * DD + col] = (bf16_t)v;
        } else {
          const int b = row >> 11, s = row & (SS - 1);
          Vt[((size_t)b * DD + col) * SS + s] = (bf16_t)v;  // transposed V
        }
      }
    }
}

// ---------------- 3) causal score GEMM (massively parallel) ----------------
// P[q,k] = (Q . K) / 32, masked to -1e9 for k>q.  Wave -> 32x64 tile;
// fully-masked tiles exit immediately.  Written region per 32-row block is
// rounded up to a 64-col boundary (masked values -1e9 pad it).
__global__ void __launch_bounds__(256) scores_gemm(
    const bf16_t* __restrict__ Qb, const bf16_t* __restrict__ Kb,
    bf16_t* __restrict__ P) {
  const int lane = threadIdx.x & 31;
  const int wid  = blockIdx.x * 8 + (threadIdx.x >> 5);
  // 4 batch * 64 row-blocks(32) * 32 col-blocks(64) = 8192 wave tasks
  const int cb = wid & 31;
  const int rb = (wid >> 5) & 63;
  const int b  = wid >> 11;
  const int q0 = rb * 32, k0 = cb * 64;
  if (k0 > q0 + 31) return;                      // fully above the diagonal

  const int lr = lane & 15, lh = lane >> 4;
  const bf16_t* aptr = Qb + ((size_t)b * SS + q0 + lr) * DD + lh * 16;
  const bf16_t* bptr = Kb + ((size_t)b * SS + k0 + lr) * DD + lh * 16;

  v8f acc[2][4] = {};
  #pragma unroll 4
  for (int kk = 0; kk < DD; kk += 32) {
    v16bf a0 = *(const v16bf*)(aptr + kk);
    v16bf a1 = *(const v16bf*)(aptr + (size_t)16 * DD + kk);
    __builtin_prefetch(aptr + kk + 128, 0, 3);   // WGP-scope prefetch
    #pragma unroll
    for (int j = 0; j < 4; ++j) {
      v16bf kf = *(const v16bf*)(bptr + (size_t)j * 16 * DD + kk);
      acc[0][j] = __builtin_amdgcn_wmma_f32_16x16x32_bf16(
          false, a0, false, kf, (short)0, acc[0][j], false, false);
      acc[1][j] = __builtin_amdgcn_wmma_f32_16x16x32_bf16(
          false, a1, false, kf, (short)0, acc[1][j], false, false);
    }
  }

  const float sc = 0.03125f;  // 1/sqrt(1024)
  #pragma unroll
  for (int mi = 0; mi < 2; ++mi)
    #pragma unroll
    for (int j = 0; j < 4; ++j) {
      const int col = k0 + j * 16 + lr;
      #pragma unroll
      for (int i = 0; i < 8; ++i) {
        const int row = q0 + mi * 16 + i + 8 * lh;
        float s = acc[mi][j][i] * sc;
        if (col > row) s = -1.0e9f;              // causal mask
        P[((size_t)b * SS + row) * SS + col] = (bf16_t)s;
      }
    }
}

// ---------------- 4) per-row softmax stats (max, sum of exp) ---------------
// One wave per row; two passes over the L2-resident bf16 score row.
__global__ void __launch_bounds__(256) row_stats(
    const bf16_t* __restrict__ P, float* __restrict__ mbuf,
    float* __restrict__ lbuf) {
  const int lane = threadIdx.x & 31;
  const int wid  = blockIdx.x * 8 + (threadIdx.x >> 5);   // row = b*S + q
  const int q    = wid & (SS - 1);
  // written region for this row's 32-block, rounded up to 64 cols
  const int nn = (((q >> 5) << 5) + 32 + 63) & ~63;
  const bf16_t* row = P + (size_t)wid * SS;

  float m = -3.0e38f;
  for (int k = lane * 2; k < nn; k += 64) {
    v2bf v = *(const v2bf*)(row + k);
    m = fmaxf(m, fmaxf((float)v[0], (float)v[1]));
  }
  m = fmaxf(m, __shfl_xor(m, 1, 32));
  m = fmaxf(m, __shfl_xor(m, 2, 32));
  m = fmaxf(m, __shfl_xor(m, 4, 32));
  m = fmaxf(m, __shfl_xor(m, 8, 32));
  m = fmaxf(m, __shfl_xor(m, 16, 32));

  float l = 0.f;
  for (int k = lane * 2; k < nn; k += 64) {
    v2bf v = *(const v2bf*)(row + k);
    l += __expf((float)v[0] - m) + __expf((float)v[1] - m);
  }
  l += __shfl_xor(l, 1, 32);
  l += __shfl_xor(l, 2, 32);
  l += __shfl_xor(l, 4, 32);
  l += __shfl_xor(l, 8, 32);
  l += __shfl_xor(l, 16, 32);

  if (lane == 0) { mbuf[wid] = m; lbuf[wid] = l; }
}

// ------------------- 5) in-place exp-normalize (L2-resident) ---------------
// One thread per 8 consecutive k (16B vector load/store, one m/l fetch).
__global__ void attn_norm(bf16_t* __restrict__ P,
                          const float* __restrict__ mbuf,
                          const float* __restrict__ lbuf) {
  const size_t t    = (size_t)blockIdx.x * blockDim.x + threadIdx.x;
  const size_t rowi = t >> 8;                     // 256 chunks of 8 per row
  const int    kc   = (int)(t & 255) * 8;
  const int    q    = (int)(rowi & (SS - 1));
  bf16_t* p = P + rowi * SS + kc;
  v8bf out;
  if (kc > q) {
    #pragma unroll
    for (int e = 0; e < 8; ++e) out[e] = (bf16_t)0.f;   // zero masked chunk
  } else {
    const float m = mbuf[rowi], rli = 1.0f / lbuf[rowi];
    const v8bf v = *(const v8bf*)p;
    #pragma unroll
    for (int e = 0; e < 8; ++e) {
      const float pe = (kc + e > q) ? 0.f : __expf((float)v[e] - m) * rli;
      out[e] = (bf16_t)pe;
    }
  }
  *(v8bf*)p = out;
}

// --------------------------- 6) P * V NT GEMM ------------------------------
// out[m,n] = sum_k P[m,k] * Vt[n,k].  Wave -> 32x64 tile, causal K bound
// (normalize zeroed k>q, so kend = m0+32 is exact).
__global__ void __launch_bounds__(256) pv_gemm(
    const bf16_t* __restrict__ P, const bf16_t* __restrict__ Vt,
    float* __restrict__ out) {
  const int lane = threadIdx.x & 31;
  const int wid  = blockIdx.x * 8 + (threadIdx.x >> 5);
  // 4 batch * 64 row-blocks(32) * 16 col-groups(64) = 4096 wave tasks
  const int ng = wid & 15;
  const int rb = (wid >> 4) & 63;
  const int b  = wid >> 10;
  const int m0 = rb * 32, n0 = ng * 64;
  const int lr = lane & 15, lh = lane >> 4;

  const bf16_t* aptr = P  + ((size_t)b * SS + m0 + lr) * SS + lh * 16;
  const bf16_t* bptr = Vt + ((size_t)b * DD + n0 + lr) * SS + lh * 16;
  const int kend = m0 + 32;

  v8f acc[2][4] = {};
  for (int kk = 0; kk < kend; kk += 32) {
    v16bf a0 = *(const v16bf*)(aptr + kk);
    v16bf a1 = *(const v16bf*)(aptr + (size_t)16 * SS + kk);
    __builtin_prefetch(aptr + kk + 128, 0, 3);   // WGP-scope prefetch
    #pragma unroll
    for (int j = 0; j < 4; ++j) {
      v16bf vf = *(const v16bf*)(bptr + (size_t)j * 16 * SS + kk);
      acc[0][j] = __builtin_amdgcn_wmma_f32_16x16x32_bf16(
          false, a0, false, vf, (short)0, acc[0][j], false, false);
      acc[1][j] = __builtin_amdgcn_wmma_f32_16x16x32_bf16(
          false, a1, false, vf, (short)0, acc[1][j], false, false);
    }
  }
  #pragma unroll
  for (int mi = 0; mi < 2; ++mi)
    #pragma unroll
    for (int j = 0; j < 4; ++j)
      #pragma unroll
      for (int i = 0; i < 8; ++i) {
        const int row = m0 + mi * 16 + i + 8 * lh;
        const int col = n0 + j * 16 + lr;
        out[((size_t)b * SS + row) * DD + col] = acc[mi][j][i];
      }
}

// ---------------------------------------------------------------------------
extern "C" void kernel_launch(void* const* d_in, const int* in_sizes, int n_in,
                              void* d_out, int out_size, void* d_ws,
                              size_t ws_size, hipStream_t stream) {
  const float* x  = (const float*)d_in[0];
  const float* Wq = (const float*)d_in[1];
  const float* bq = (const float*)d_in[2];
  const float* Wk = (const float*)d_in[3];
  const float* bk = (const float*)d_in[4];
  const float* Wv = (const float*)d_in[5];
  const float* bv = (const float*)d_in[6];
  // d_in[7] = mask: causal tril by construction, enforced analytically.
  float* out = (float*)d_out;

  char* ws = (char*)d_ws;
  size_t off = 0;
  bf16_t* xb  = (bf16_t*)(ws + off); off += (size_t)BB * SS * DD * 2;  // 16MB
  bf16_t* Wqb = (bf16_t*)(ws + off); off += (size_t)DD * DD * 2;       //  2MB
  bf16_t* Wkb = (bf16_t*)(ws + off); off += (size_t)DD * DD * 2;
  bf16_t* Wvb = (bf16_t*)(ws + off); off += (size_t)DD * DD * 2;
  bf16_t* Qb  = (bf16_t*)(ws + off); off += (size_t)BB * SS * DD * 2;  // 16MB
  bf16_t* Kb  = (bf16_t*)(ws + off); off += (size_t)BB * SS * DD * 2;
  bf16_t* Vt  = (bf16_t*)(ws + off); off += (size_t)BB * SS * DD * 2;
  bf16_t* P   = (bf16_t*)(ws + off); off += (size_t)BB * SS * SS * 2;  // 32MB
  float*  mbuf = (float*)(ws + off); off += (size_t)BB * SS * 4;
  float*  lbuf = (float*)(ws + off); off += (size_t)BB * SS * 4;

  // 1) convert activations + weights to bf16
  cvt_f32_bf16<<<(BB * SS * DD + 255) / 256, 256, 0, stream>>>(x, xb, BB * SS * DD);
  cvt_f32_bf16<<<(DD * DD + 255) / 256, 256, 0, stream>>>(Wq, Wqb, DD * DD);
  cvt_f32_bf16<<<(DD * DD + 255) / 256, 256, 0, stream>>>(Wk, Wkb, DD * DD);
  cvt_f32_bf16<<<(DD * DD + 255) / 256, 256, 0, stream>>>(Wv, Wvb, DD * DD);

  // 2) Q,K,V projections (V stored transposed): 12288 waves / 8 per block
  qkv_gemm<<<1536, 256, 0, stream>>>(xb, Wqb, Wkb, Wvb, bq, bk, bv, Qb, Kb, Vt);

  // 3) causal score GEMM: 8192 wave tasks (upper triangle exits early)
  scores_gemm<<<1024, 256, 0, stream>>>(Qb, Kb, P);

  // 4) per-row softmax stats: one wave per row, 8192 waves
  row_stats<<<1024, 256, 0, stream>>>(P, mbuf, lbuf);

  // 5) normalize probabilities in place (P is L2-resident: 32MB < 192MB)
  attn_norm<<<(BB * SS * SS / 8 + 255) / 256, 256, 0, stream>>>(P, mbuf, lbuf);

  // 6) out = P @ V : 4096 wave tasks
  pv_gemm<<<512, 256, 0, stream>>>(P, Vt, out);
}
